// ConvLSTMModel_90486370992845
// MI455X (gfx1250) — compile-verified, matrix operand order checked by reading, not run on hardware
//
#include <hip/hip_runtime.h>
#include <hip/hip_bf16.h>
#include <math.h>

// ---------------------------------------------------------------------------
// CDNA5 (gfx1250) ConvLSTM stack: bf16 WMMA implicit GEMM with TDM-staged
// A-operand in LDS (tensor_load_to_lds + s_wait_tensorcnt double buffering).
// ---------------------------------------------------------------------------

typedef __attribute__((ext_vector_type(16))) __bf16 v16bf;
typedef __attribute__((ext_vector_type(8)))  float  v8f;
typedef __attribute__((ext_vector_type(4)))  unsigned int u32x4;
typedef __attribute__((ext_vector_type(8)))  unsigned int u32x8;

#define BATCH 2
#define TSTEPS 8
#define KCHUNK 32                 // k-blocks staged per LDS buffer (32 KiB)
#define CHUNK_BYTES (KCHUNK * 1024)

__device__ __forceinline__ unsigned short f2bf(float f) {
  unsigned int u = __float_as_uint(f);
  u += 0x7FFFu + ((u >> 16) & 1u);          // round-to-nearest-even
  return (unsigned short)(u >> 16);
}

__device__ __forceinline__ float hsig(float x) {
  return fminf(fmaxf(0.2f * x + 0.5f, 0.0f), 1.0f);
}

// K index inside a 32-wide K block for (lane, half-element e), per ISA 7.12.2
__device__ __forceinline__ int kInBlock(int lane, int e) {
  return e + ((e >= 8) ? 8 : 0) + ((lane >= 16) ? 8 : 0);
}

// ---------------------------------------------------------------------------
// 1-D TDM copy global->LDS: one line of (bytes/8) 8-byte elements.
// D# group0/group1 packing per CDNA5 ISA.
// ---------------------------------------------------------------------------
__device__ __forceinline__ void tdm_load_1d(unsigned int ldsOff,
                                            const void* gsrc,
                                            unsigned int bytes) {
  unsigned long long ga = (unsigned long long)(uintptr_t)gsrc;
  unsigned int d0 = bytes >> 3;  // elements of 8 bytes
  u32x4 g0;
  g0[0] = 1u;                                            // count=1, user mode
  g0[1] = ldsOff;                                        // lds_addr
  g0[2] = (unsigned int)ga;                              // global_addr[31:0]
  g0[3] = ((unsigned int)(ga >> 32) & 0x1FFFFFFu)        // global_addr[56:32]
          | (2u << 30);                                  // type=2 ("image")
  u32x8 g1;
  g1[0] = 3u << 16;                                      // data_size=8B
  g1[1] = (d0 & 0xFFFFu) << 16;                          // tensor_dim0[15:0]
  g1[2] = (d0 >> 16) | (1u << 16);                       // dim0 hi, tensor_dim1=1
  g1[3] = (d0 & 0xFFFFu) << 16;                          // tile_dim0 = d0
  g1[4] = 0u;                                            // tile_dim1/2 unused
  g1[5] = d0;                                            // tensor_dim0_stride
  g1[6] = 0u;
  g1[7] = 0u;
  asm volatile("tensor_load_to_lds %0, %1" :: "s"(g0), "s"(g1) : "memory");
}

// ---------------------------------------------------------------------------
// Pack [k ; rk] into WMMA B-fragment order: [nb][kb][lane][e] (bf16).
// ---------------------------------------------------------------------------
__global__ void pack_weights_kernel(const float* __restrict__ k,
                                    const float* __restrict__ rk,
                                    unsigned short* __restrict__ wp,
                                    int Cin, int F, int N4, int Kdim,
                                    int kBlocks, int nBlocks) {
  int id = blockIdx.x * blockDim.x + threadIdx.x;
  int total = nBlocks * kBlocks * 32;
  if (id >= total) return;
  int lane = id & 31;
  int kb = (id >> 5) % kBlocks;
  int nb = id / (32 * kBlocks);
  int n = nb * 16 + (lane & 15);
  int Csum = Cin + F;
  unsigned short* dst = wp + ((size_t)(nb * kBlocks + kb) * 32 + lane) * 16;
#pragma unroll
  for (int e = 0; e < 16; ++e) {
    int K = kb * 32 + kInBlock(lane, e);
    float v = 0.0f;
    if (K < Kdim) {
      int tap = K / Csum;
      int ch  = K - tap * Csum;
      if (ch < Cin) v = k[((size_t)(tap * Cin + ch)) * N4 + n];
      else          v = rk[((size_t)(tap * F + (ch - Cin))) * N4 + n];
    }
    dst[e] = f2bf(v);
  }
}

// ---------------------------------------------------------------------------
// im2col into WMMA A-fragment order: [b][mb][kb][lane][e] (bf16).
// ---------------------------------------------------------------------------
__global__ void im2col_kernel(const float* __restrict__ xin,
                              const float* __restrict__ hout,
                              unsigned short* __restrict__ ap,
                              int t, int H, int W, int Cin, int F,
                              int Kdim, int kBlocks, int mBlocks) {
  int id = blockIdx.x * blockDim.x + threadIdx.x;
  int total = BATCH * mBlocks * kBlocks * 32;
  if (id >= total) return;
  int lane = id & 31;
  int kb = (id >> 5) % kBlocks;
  int mb = (id / (32 * kBlocks)) % mBlocks;
  int b  = id / (32 * kBlocks * mBlocks);
  int m = mb * 16 + (lane & 15);
  int y = m / W, x = m % W;
  int Csum = Cin + F;
  unsigned short* dst =
      ap + ((size_t)((b * mBlocks + mb) * kBlocks + kb) * 32 + lane) * 16;
#pragma unroll
  for (int e = 0; e < 16; ++e) {
    int K = kb * 32 + kInBlock(lane, e);
    float v = 0.0f;
    if (K < Kdim) {
      int tap = K / Csum;
      int ch  = K - tap * Csum;
      int yy = y + tap / 3 - 1;
      int xx = x + tap % 3 - 1;
      if (yy >= 0 && yy < H && xx >= 0 && xx < W) {
        if (ch < Cin) {
          v = xin[(((size_t)(b * TSTEPS + t) * H + yy) * W + xx) * Cin + ch];
        } else if (t > 0) {
          v = hout[(((size_t)(b * TSTEPS + t - 1) * H + yy) * W + xx) * F +
                   (ch - Cin)];
        }
      }
    }
    dst[e] = f2bf(v);
  }
}

// ---------------------------------------------------------------------------
// GEMM: 256-thread block (8 waves). Block covers one 16-row m-tile x 512
// columns (8 waves x 4 n-tiles). A fragments for a K-chunk are staged into
// LDS by the Tensor Data Mover (double buffered); B streams from L2.
// ---------------------------------------------------------------------------
__global__ void __launch_bounds__(256) gemm_wmma_kernel(
    const unsigned short* __restrict__ ap, const unsigned short* __restrict__ wp,
    const float* __restrict__ bias, float* __restrict__ z,
    int mBlocks, int kBlocks, int nChunks, int N4, int M) {
  extern __shared__ unsigned short smem[];   // 2 * CHUNK_BYTES
  int lane = threadIdx.x & 31;
  int wave = threadIdx.x >> 5;               // 0..7
  int nc = blockIdx.x % nChunks;
  int mb = (blockIdx.x / nChunks) % mBlocks;
  int b  = blockIdx.x / (nChunks * mBlocks);
  int nb0 = nc * 32 + wave * 4;              // 4 n-tiles per wave

  v8f acc[4];
#pragma unroll
  for (int j = 0; j < 4; ++j)
#pragma unroll
    for (int r = 0; r < 8; ++r) acc[j][r] = 0.0f;

  const char* aG =
      (const char*)ap + (size_t)(b * mBlocks + mb) * kBlocks * 1024;
  unsigned int ldsBase = (unsigned int)(uintptr_t)smem;  // LDS byte offset
  int nkc = (kBlocks + KCHUNK - 1) / KCHUNK;

  if (wave == 0) {
    int kcnt0 = (kBlocks < KCHUNK) ? kBlocks : KCHUNK;
    tdm_load_1d(ldsBase, aG, (unsigned int)kcnt0 * 1024u);
    __builtin_amdgcn_s_wait_tensorcnt(0);
  }
  __syncthreads();

  for (int c = 0; c < nkc; ++c) {
    int kb0 = c * KCHUNK;
    int kcnt = kBlocks - kb0;
    if (kcnt > KCHUNK) kcnt = KCHUNK;
    int p = c & 1;
    if (wave == 0 && c + 1 < nkc) {            // prefetch next chunk via TDM
      int kb1 = kb0 + KCHUNK;
      int knext = kBlocks - kb1;
      if (knext > KCHUNK) knext = KCHUNK;
      tdm_load_1d(ldsBase + (unsigned int)(1 - p) * CHUNK_BYTES,
                  aG + (size_t)kb1 * 1024, (unsigned int)knext * 1024u);
    }
    const unsigned short* aL = smem + (size_t)p * (CHUNK_BYTES / 2);
    for (int kk = 0; kk < kcnt; ++kk) {
      v16bf afrag =
          *reinterpret_cast<const v16bf*>(aL + (size_t)kk * 512 + lane * 16);
      int kb = kb0 + kk;
#pragma unroll
      for (int j = 0; j < 4; ++j) {
        const unsigned short* bPtr =
            wp + (((size_t)(nb0 + j) * kBlocks + kb) * 32 + lane) * 16;
        v16bf bfrag = *reinterpret_cast<const v16bf*>(bPtr);
        acc[j] = __builtin_amdgcn_wmma_f32_16x16x32_bf16(
            false, afrag, false, bfrag, (short)0, acc[j], false, false);
      }
    }
    if (wave == 0 && c + 1 < nkc) __builtin_amdgcn_s_wait_tensorcnt(0);
    __syncthreads();
  }

  int nlo  = lane & 15;
  int mrow = mb * 16 + ((lane >= 16) ? 8 : 0);
#pragma unroll
  for (int j = 0; j < 4; ++j) {
    int n = (nb0 + j) * 16 + nlo;
    float bv = bias[n];
#pragma unroll
    for (int r = 0; r < 8; ++r) {
      z[((size_t)b * M + (mrow + r)) * N4 + n] = acc[j][r] + bv;
    }
  }
}

// ---------------------------------------------------------------------------
// Instance-norm statistics: per (b, f) mean and rsqrt(var+eps) over M pixels.
// ---------------------------------------------------------------------------
__global__ void inorm_stats_kernel(const float* __restrict__ src,
                                   float* __restrict__ mean,
                                   float* __restrict__ rstd,
                                   int M, int C, int off, int F) {
  int b = blockIdx.x / F;
  int f = blockIdx.x % F;
  float s = 0.0f, ss = 0.0f;
  for (int m = threadIdx.x; m < M; m += blockDim.x) {
    float v = src[((size_t)b * M + m) * C + off + f];
    s += v;
    ss += v * v;
  }
  __shared__ float sh[256], sh2[256];
  sh[threadIdx.x] = s;
  sh2[threadIdx.x] = ss;
  __syncthreads();
  for (int st = blockDim.x >> 1; st > 0; st >>= 1) {
    if ((int)threadIdx.x < st) {
      sh[threadIdx.x]  += sh[threadIdx.x + st];
      sh2[threadIdx.x] += sh2[threadIdx.x + st];
    }
    __syncthreads();
  }
  if (threadIdx.x == 0) {
    float mu  = sh[0] / (float)M;
    float var = sh2[0] / (float)M - mu * mu;
    mean[blockIdx.x] = mu;
    rstd[blockIdx.x] = rsqrtf(var + 0.001f);
  }
}

// ---------------------------------------------------------------------------
// Gates: c_new = hsig(zf)*c_old + hsig(zi)*tanh(inorm(zc)); stash o-gate.
// ---------------------------------------------------------------------------
__global__ void gates_kernel(const float* __restrict__ z, float* __restrict__ c,
                             float* __restrict__ o,
                             const float* __restrict__ mean,
                             const float* __restrict__ rstd,
                             const float* __restrict__ g,
                             const float* __restrict__ bt,
                             int M, int F, int t) {
  int id = blockIdx.x * blockDim.x + threadIdx.x;
  int total = BATCH * M * F;
  if (id >= total) return;
  int f = id % F;
  int b = id / (F * M);
  const float* zr = z + ((size_t)(id / F)) * 4 * F;  // (b*M+m)*4F
  float ig = hsig(zr[f]);
  float fg = hsig(zr[F + f]);
  float og = hsig(zr[3 * F + f]);
  float zn = (zr[2 * F + f] - mean[b * F + f]) * rstd[b * F + f] * g[f] + bt[f];
  float cold = (t > 0) ? c[id] : 0.0f;
  c[id] = fg * cold + ig * tanhf(zn);
  o[id] = og;
}

// ---------------------------------------------------------------------------
// h_t = o * tanh(inorm(c_new)); write into layer output (B,T,M,F) at time t.
// ---------------------------------------------------------------------------
__global__ void hout_kernel(const float* __restrict__ c,
                            const float* __restrict__ o,
                            const float* __restrict__ mean,
                            const float* __restrict__ rstd,
                            const float* __restrict__ g,
                            const float* __restrict__ bt,
                            float* __restrict__ hout, int M, int F, int t) {
  int id = blockIdx.x * blockDim.x + threadIdx.x;
  int total = BATCH * M * F;
  if (id >= total) return;
  int f = id % F;
  int m = (id / F) % M;
  int b = id / (F * M);
  float cn = (c[id] - mean[b * F + f]) * rstd[b * F + f] * g[f] + bt[f];
  hout[(((size_t)b * TSTEPS + t) * M + m) * F + f] = o[id] * tanhf(cn);
}

// ---------------------------------------------------------------------------
// 2x2 stride-2 max pool over H,W: (B,T,H,W,F) -> (B,T,H/2,W/2,F).
// ---------------------------------------------------------------------------
__global__ void maxpool_kernel(const float* __restrict__ in,
                               float* __restrict__ out, int H, int W, int F) {
  int H2 = H >> 1, W2 = W >> 1;
  long long id = (long long)blockIdx.x * blockDim.x + threadIdx.x;
  long long total = (long long)BATCH * TSTEPS * H2 * W2 * F;
  if (id >= total) return;
  int f  = id % F;
  int x2 = (id / F) % W2;
  int y2 = (id / ((long long)F * W2)) % H2;
  long long bt = id / ((long long)F * W2 * H2);  // b*T + t
  size_t base = ((bt * H + 2 * y2) * W + 2 * x2) * (size_t)F + f;
  size_t rs = (size_t)W * F;
  float v = fmaxf(fmaxf(in[base], in[base + F]),
                  fmaxf(in[base + rs], in[base + rs + F]));
  out[id] = v;
}

// ---------------------------------------------------------------------------
// Head: mean over 4x4 of h5[:, T-1], dense 512->10, softmax. 1 block / batch.
// ---------------------------------------------------------------------------
__global__ void head_kernel(const float* __restrict__ h5,
                            const float* __restrict__ Wd,
                            const float* __restrict__ bd,
                            float* __restrict__ out, int M, int F) {
  __shared__ float pooled[512];
  __shared__ float logits[10];
  int b = blockIdx.x;
  int f = threadIdx.x;
  if (f < F) {
    float s = 0.0f;
    for (int m = 0; m < M; ++m)
      s += h5[(((size_t)b * TSTEPS + (TSTEPS - 1)) * M + m) * F + f];
    pooled[f] = s / (float)M;
  }
  __syncthreads();
  if (f < 10) {
    float s = bd[f];
    for (int c = 0; c < F; ++c) s += pooled[c] * Wd[c * 10 + f];
    logits[f] = s;
  }
  __syncthreads();
  if (f == 0) {
    float mx = logits[0];
    for (int j = 1; j < 10; ++j) mx = fmaxf(mx, logits[j]);
    float e[10], s = 0.0f;
    for (int j = 0; j < 10; ++j) { e[j] = expf(logits[j] - mx); s += e[j]; }
    for (int j = 0; j < 10; ++j) out[b * 10 + j] = e[j] / s;
  }
}

// ---------------------------------------------------------------------------
// Host orchestration
// ---------------------------------------------------------------------------
extern "C" void kernel_launch(void* const* d_in, const int* in_sizes, int n_in,
                              void* d_out, int out_size, void* d_ws,
                              size_t ws_size, hipStream_t stream) {
  (void)in_sizes; (void)n_in; (void)out_size; (void)ws_size;
  const float* x = (const float*)d_in[0];
  const float* Wd = (const float*)d_in[26];
  const float* bd = (const float*)d_in[27];
  float* out = (float*)d_out;

  char* ws = (char*)d_ws;
  float* actA           = (float*)(ws);                               //  8 MiB
  float* actB           = (float*)(ws + ((size_t)8  << 20));          //  8 MiB
  unsigned short* wpack = (unsigned short*)(ws + ((size_t)16 << 20)); // 36 MiB
  unsigned short* apack = (unsigned short*)(ws + ((size_t)53 << 20)); //  9 MiB
  float* zbuf           = (float*)(ws + ((size_t)62 << 20));          //  4 MiB
  float* cbuf           = (float*)(ws + ((size_t)66 << 20));          //  1 MiB
  float* obuf           = (float*)(ws + ((size_t)67 << 20));          //  1 MiB
  float* meanb          = (float*)(ws + ((size_t)68 << 20));
  float* rstdb          = (float*)(ws + ((size_t)68 << 20) + 4096);

  struct { int Cin, F, H; } cfg[5] = {
      {3, 128, 32}, {128, 128, 32}, {128, 512, 16}, {512, 512, 8}, {512, 512, 4}};

  const float* inPtr = x;
  for (int L = 0; L < 5; ++L) {
    int Cin = cfg[L].Cin, F = cfg[L].F, H = cfg[L].H, W = H;
    int M = H * W, N4 = 4 * F;
    int Kdim = 9 * (Cin + F);
    int kB = (Kdim + 31) / 32;
    int mB = M / 16;
    int nB = N4 / 16;
    int nChunks = nB / 32;                 // 512-column chunks per block
    const float* k    = (const float*)d_in[1 + 5 * L + 0];
    const float* rk   = (const float*)d_in[1 + 5 * L + 1];
    const float* bias = (const float*)d_in[1 + 5 * L + 2];
    const float* g    = (const float*)d_in[1 + 5 * L + 3];
    const float* bt   = (const float*)d_in[1 + 5 * L + 4];
    float* outPtr = (L == 0) ? actA : actB;

    {
      int tot = nB * kB * 32;
      pack_weights_kernel<<<(tot + 255) / 256, 256, 0, stream>>>(
          k, rk, wpack, Cin, F, N4, Kdim, kB, nB);
    }

    for (int t = 0; t < TSTEPS; ++t) {
      {
        int tot = BATCH * mB * kB * 32;
        im2col_kernel<<<(tot + 255) / 256, 256, 0, stream>>>(
            inPtr, outPtr, apack, t, H, W, Cin, F, Kdim, kB, mB);
      }
      {
        int blocks = BATCH * mB * nChunks;
        size_t lds = 2 * CHUNK_BYTES;
        gemm_wmma_kernel<<<blocks, 256, lds, stream>>>(apack, wpack, bias,
                                                       zbuf, mB, kB, nChunks,
                                                       N4, M);
      }
      inorm_stats_kernel<<<BATCH * F, 256, 0, stream>>>(zbuf, meanb, rstdb, M,
                                                        N4, 2 * F, F);
      {
        int tot = BATCH * M * F;
        gates_kernel<<<(tot + 255) / 256, 256, 0, stream>>>(
            zbuf, cbuf, obuf, meanb, rstdb, g, bt, M, F, t);
      }
      inorm_stats_kernel<<<BATCH * F, 256, 0, stream>>>(cbuf, meanb, rstdb, M,
                                                        F, 0, F);
      {
        int tot = BATCH * M * F;
        hout_kernel<<<(tot + 255) / 256, 256, 0, stream>>>(
            cbuf, obuf, meanb, rstdb, g, bt, outPtr, M, F, t);
      }
    }

    if (L >= 1 && L <= 3) {  // maxpool after layers 2,3,4
      long long tot = (long long)BATCH * TSTEPS * (H / 2) * (W / 2) * F;
      maxpool_kernel<<<(int)((tot + 255) / 256), 256, 0, stream>>>(outPtr, actA,
                                                                   H, W, F);
      inPtr = actA;
    } else if (L == 0) {
      inPtr = actA;
    }
  }

  head_kernel<<<BATCH, 512, 0, stream>>>(actB, Wd, bd, out, 16, 512);
}
